// topk_LeNet_20753281974404
// MI455X (gfx1250) — compile-verified, matrix operand order checked by reading, not run on hardware
//
#include <hip/hip_runtime.h>
#include <hip/hip_bf16.h>

// ---------- types ----------
typedef __bf16 bf16_t;
typedef __attribute__((ext_vector_type(16))) __bf16 v16bf;
typedef __attribute__((ext_vector_type(8)))  __bf16 v8bf;
typedef __attribute__((ext_vector_type(8)))  float  v8f;
typedef __attribute__((ext_vector_type(4)))  unsigned int v4u;
typedef __attribute__((ext_vector_type(8)))  int v8i_;
typedef __attribute__((ext_vector_type(4)))  int v4i_;

#if defined(__has_builtin)
#if __has_builtin(__builtin_amdgcn_tensor_load_to_lds) && \
    __has_builtin(__builtin_amdgcn_s_wait_tensorcnt)
#define FC1_USE_TDM 1
#endif
#endif

// Load one WMMA 16-bit fragment slice for this lane: 8 contiguous bf16 at p
// (K = k0..k0+7) and 8 at p+16 (K = k0+16..k0+23).  Caller folds the
// lane>=16 "+8" K offset into p.
__device__ inline v16bf load_frag(const bf16_t* p) {
  v8bf lo = *(const v8bf*)(p);
  v8bf hi = *(const v8bf*)(p + 16);
  v16bf r;
#pragma unroll
  for (int i = 0; i < 8; ++i) { r[i] = lo[i]; r[i + 8] = hi[i]; }
  return r;
}

// ---------- tiny helpers ----------
__global__ void zero_pad_kernel(bf16_t* z) {
  if (threadIdx.x < 64) z[threadIdx.x] = (bf16_t)0.0f;   // 128B of zeros
}

// conv2_w OIHW [64][32][5][5] fp32  ->  bf16 [25][64][32]  (tap-major, cout, cin)
__global__ void cvt_w2_kernel(const float* __restrict__ w2, bf16_t* __restrict__ w2b) {
  int tid = blockIdx.x * blockDim.x + threadIdx.x;
  if (tid >= 25 * 64 * 32) return;
  int ci = tid & 31;
  int co = (tid >> 5) & 63;
  int rs = tid >> 11;
  w2b[tid] = (bf16_t)w2[(co * 32 + ci) * 25 + rs];
}

// fc1_w [1024][3136] fp32 -> bf16 same layout
__global__ void cvt_w1_kernel(const float* __restrict__ w, bf16_t* __restrict__ o, int n) {
  int tid = blockIdx.x * blockDim.x + threadIdx.x;
  if (tid < n) o[tid] = (bf16_t)w[tid];
}

// ---------- stage 1: conv1 + top-k + trelu + maxpool -> NHWC bf16 [B,14,14,32] ----------
__global__ void conv1_topk_pool_kernel(const float* __restrict__ x,
                                       const float* __restrict__ w1,
                                       const float* __restrict__ tr,
                                       bf16_t* __restrict__ pooled1,
                                       int total /* B*14*14 */) {
  int tid = blockIdx.x * blockDim.x + threadIdx.x;
  if (tid >= total) return;
  int pw = tid % 14;
  int t  = tid / 14;
  int ph = t % 14;
  int b  = t / 14;
  int h0 = 2 * ph, w0 = 2 * pw;

  // 6x6 input patch covering the 2x2 pooled window of 5x5 convs (pad 2)
  float patch[36];
  const float* xb = x + b * 784;
#pragma unroll
  for (int r = 0; r < 6; ++r) {
    int ih = h0 + r - 2;
#pragma unroll
    for (int s = 0; s < 6; ++s) {
      int iw = w0 + s - 2;
      bool ok = (ih >= 0) & (ih < 28) & (iw >= 0) & (iw < 28);
      patch[r * 6 + s] = ok ? xb[ih * 28 + iw] : 0.0f;
    }
  }

  // pass 1: running top-4 per sub-pixel (value semantics match xt >= kth incl. ties)
  float a0[4], a1[4], a2[4], a3[4];
#pragma unroll
  for (int sp = 0; sp < 4; ++sp) { a0[sp] = a1[sp] = a2[sp] = a3[sp] = -3.4e38f; }

  for (int c = 0; c < 32; ++c) {
    const float* wc = w1 + c * 25;
    float v[4] = {0.f, 0.f, 0.f, 0.f};
#pragma unroll
    for (int r = 0; r < 5; ++r)
#pragma unroll
      for (int s = 0; s < 5; ++s) {
        float wv = wc[r * 5 + s];
        v[0] = fmaf(patch[r * 6 + s],           wv, v[0]);
        v[1] = fmaf(patch[r * 6 + s + 1],       wv, v[1]);
        v[2] = fmaf(patch[(r + 1) * 6 + s],     wv, v[2]);
        v[3] = fmaf(patch[(r + 1) * 6 + s + 1], wv, v[3]);
      }
#pragma unroll
    for (int sp = 0; sp < 4; ++sp) {
      float val = v[sp];
      if (val > a0[sp])      { a3[sp] = a2[sp]; a2[sp] = a1[sp]; a1[sp] = a0[sp]; a0[sp] = val; }
      else if (val > a1[sp]) { a3[sp] = a2[sp]; a2[sp] = a1[sp]; a1[sp] = val; }
      else if (val > a2[sp]) { a3[sp] = a2[sp]; a2[sp] = val; }
      else if (val > a3[sp]) { a3[sp] = val; }
    }
  }

  // pass 2: recompute, mask by 4th-largest, trelu, max-pool, store NHWC bf16
  bf16_t* outp = pooled1 + (((b * 14 + ph) * 14 + pw) << 5);
  for (int c = 0; c < 32; ++c) {
    const float* wc = w1 + c * 25;
    float v[4] = {0.f, 0.f, 0.f, 0.f};
#pragma unroll
    for (int r = 0; r < 5; ++r)
#pragma unroll
      for (int s = 0; s < 5; ++s) {
        float wv = wc[r * 5 + s];
        v[0] = fmaf(patch[r * 6 + s],           wv, v[0]);
        v[1] = fmaf(patch[r * 6 + s + 1],       wv, v[1]);
        v[2] = fmaf(patch[(r + 1) * 6 + s],     wv, v[2]);
        v[3] = fmaf(patch[(r + 1) * 6 + s + 1], wv, v[3]);
      }
    float th = tr[c];
    float m = 0.0f;
#pragma unroll
    for (int sp = 0; sp < 4; ++sp) {
      float val = (v[sp] >= a3[sp]) ? v[sp] : 0.0f;
      val = fmaxf(val - th, 0.0f);
      m = fmaxf(m, val);
    }
    outp[c] = (bf16_t)m;
  }
}

// ---------- stage 2: conv2, one wave = 16 pixels x 64 couts, 100 WMMAs ----------
// pooled1 NHWC bf16 [B,14,14,32]; w2b [25][64][32]; zbuf = 128B zeros
// out NHWC bf16 [B*196, 64]
__global__ __launch_bounds__(256) void conv2_wmma_kernel(
    const bf16_t* __restrict__ pooled1, const bf16_t* __restrict__ w2b,
    const float* __restrict__ bias, const bf16_t* __restrict__ zbuf,
    bf16_t* __restrict__ out) {
  int lane = threadIdx.x & 31;
  int wave = threadIdx.x >> 5;
  int mt = blockIdx.x * 8 + wave;         // one 16-pixel M tile per wave
  int l15  = lane & 15;
  int koff = (lane >= 16) ? 8 : 0;

  int p = mt * 16 + l15;                  // this lane's A row (pixel)
  int b = p / 196;
  int rem = p % 196;
  int h = rem / 14;
  int w = rem % 14;

  v8f acc[4] = {v8f{}, v8f{}, v8f{}, v8f{}};
#pragma unroll
  for (int r = 0; r < 5; ++r) {
#pragma unroll
    for (int s = 0; s < 5; ++s) {
      int ih = h + r - 2, iw = w + s - 2;
      bool ok = (ih >= 0) & (ih < 14) & (iw >= 0) & (iw < 14);
      // clamp OOB lanes to a zero buffer: uniform control flow, no exec churn
      const bf16_t* ap =
          ok ? (pooled1 + ((((b * 14 + ih) * 14 + iw) << 5) + koff)) : (zbuf + koff);
      v16bf afrag = load_frag(ap);
      const bf16_t* wp = w2b + (((r * 5 + s) << 11) + koff);  // [tap][64][32]
#pragma unroll
      for (int t = 0; t < 4; ++t) {
        v16bf bfrag = load_frag(wp + ((t * 16 + l15) << 5));
        acc[t] = __builtin_amdgcn_wmma_f32_16x16x32_bf16(
            false, afrag, false, bfrag, (short)0, acc[t], false, false);
      }
    }
  }

  int rowoff = (lane >= 16) ? 8 : 0;
#pragma unroll
  for (int t = 0; t < 4; ++t) {
    int n = t * 16 + l15;
    float bb = bias[n];
#pragma unroll
    for (int j = 0; j < 8; ++j) {
      int m = mt * 16 + j + rowoff;
      out[m * 64 + n] = (bf16_t)fmaxf(acc[t][j] + bb, 0.0f);  // bias+relu fused
    }
  }
}

// ---------- stage 3: 2x2 maxpool + flatten to [B,3136] (c*49+h*7+w order) ----------
__global__ void pool2_flatten_kernel(const bf16_t* __restrict__ cin,
                                     bf16_t* __restrict__ act, int total) {
  int tid = blockIdx.x * blockDim.x + threadIdx.x;
  if (tid >= total) return;           // total = B*64*49
  int pw = tid % 7;
  int t  = tid / 7;
  int ph = t % 7;   t /= 7;
  int c  = t % 64;
  int b  = t / 64;
  const bf16_t* base = cin + (((b * 14 + 2 * ph) * 14 + 2 * pw) * 64 + c);
  float m0 = (float)base[0];
  float m1 = (float)base[64];
  float m2 = (float)base[14 * 64];
  float m3 = (float)base[14 * 64 + 64];
  float m = fmaxf(fmaxf(m0, m1), fmaxf(m2, m3));
  act[b * 3136 + c * 49 + ph * 7 + pw] = (bf16_t)m;
}

// ---------- stage 4: fc1 GEMM  M=2048 N=1024 K=3136 ----------
// Block = 8 waves -> 128x64 output panel.  B k-slices staged in LDS,
// double-buffered.  Staging is done by the Tensor Data Mover (one descriptor
// per k-step, issued by wave 0, tracked on TENSORcnt) when the TDM builtin
// exists; otherwise by cooperative vector loads.  LDS rows are padded to
// 80B (stride 40 halfs) so the 16 B-fragment ds_load_b128s hit 16 banks;
// the TDM descriptor reproduces the same padding with pad_interval=16 DWORDs
// (64B row) / pad_amount=4 DWORDs (16B).
#define FC1_LSTRIDE 40
#define FC1_BUFBYTES (64 * FC1_LSTRIDE * 2)

#if defined(FC1_USE_TDM)
// Issue one TDM 2D tile load: 64 rows x 32 bf16 from wt (row stride 3136
// elements) into LDS at byte offset lds_off, with 16B pad after each 64B row.
// NOTE: this toolchain carries the 6-arg builtin
//   (uint32x4 g0, int32x8 g1, int32x4, int32x4, int32x8, i32 cpol)
// -- groups 2/3 and the extra group are unused (zero) for a 2D tile.
__device__ inline void fc1_tdm_issue(const bf16_t* gsrc, unsigned lds_off) {
  unsigned long long ga = (unsigned long long)(const void*)gsrc;
  v4u g0;
  g0[0] = 1u;                                   // count=1, user descriptor
  g0[1] = lds_off;                              // lds_addr (bytes)
  g0[2] = (unsigned)(ga & 0xffffffffu);         // global_addr[31:0]
  g0[3] = (unsigned)((ga >> 32) & 0x01ffffffu)  // global_addr[56:32]
          | (2u << 30);                         // type=2 ("image")
  v8i_ g1;
  g1[0] = (1 << 16)      // data_size = 2 bytes
        | (1 << 20)      // pad_enable
        | (3 << 22)      // pad_interval: 16 DWORDs (64B) between pads
        | (3 << 25);     // pad_amount: 4 DWORDs (16B)
  g1[1] = (3136 & 0xffff) << 16;                // tensor_dim0[15:0] (=3136)
  g1[2] = (3136 >> 16) | ((1024 & 0xffff) << 16);  // dim0 hi, tensor_dim1 lo
  g1[3] = (1024 >> 16) | (32 << 16);            // dim1 hi, tile_dim0 = 32
  g1[4] = 64;                                   // tile_dim1 = 64, tile_dim2 = 0
  g1[5] = 3136;                                 // tensor_dim0_stride[31:0]
  g1[6] = 0;                                    // stride hi / dim1_stride lo
  g1[7] = 0;
  v4i_ z4 = {0, 0, 0, 0};                       // 2D: groups 2/3 unused
  v8i_ z8 = {0, 0, 0, 0, 0, 0, 0, 0};
  __builtin_amdgcn_tensor_load_to_lds(g0, g1, z4, z4, z8, 0);
}
#endif

__global__ __launch_bounds__(256) void fc1_wmma_kernel(
    const bf16_t* __restrict__ act, const bf16_t* __restrict__ wt,
    const float* __restrict__ bias, float* __restrict__ out) {
  __shared__ __align__(16) bf16_t lds_b[2][64 * FC1_LSTRIDE];

  int lane = threadIdx.x & 31;
  int wave = threadIdx.x >> 5;
  int mgrp = blockIdx.x >> 4;         // 16 groups of 128 rows
  int ngrp = blockIdx.x & 15;         // 16 groups of 64 cols
  int l15  = lane & 15;
  int koff = (lane >= 16) ? 8 : 0;

  int mbase = mgrp * 128 + wave * 16;
  int nbase = ngrp * 64;
  const bf16_t* arow = act + (size_t)(mbase + l15) * 3136;

  v8f acc[4] = {v8f{}, v8f{}, v8f{}, v8f{}};

#if defined(FC1_USE_TDM)
  const bf16_t* btile = wt + (size_t)nbase * 3136;  // row stride 3136 elems
  if (threadIdx.x < 32) fc1_tdm_issue(btile, 0u);   // prologue: tile 0 -> buf 0
  for (int kk = 0; kk < 98; ++kk) {
    int k0 = kk * 32;
    bf16_t* buf = lds_b[kk & 1];
    if (threadIdx.x < 32) {
      if (kk + 1 < 98) {
        // next tile into the other buffer (its readers passed last barrier)
        fc1_tdm_issue(btile + (kk + 1) * 32, (unsigned)(((kk + 1) & 1) * FC1_BUFBYTES));
        __builtin_amdgcn_s_wait_tensorcnt(1);   // in-order: tile kk resident
      } else {
        __builtin_amdgcn_s_wait_tensorcnt(0);
      }
    }
    __syncthreads();                            // publish tile kk to all waves
    // The TDM wrote lds_b behind the compiler's back (descriptor holds a raw
    // LDS offset).  Escape the buffer address into opaque asm with a memory
    // clobber so the ds_loads below cannot be folded to undef/eliminated.
    asm volatile("" ::"r"((void*)lds_b) : "memory");
    __builtin_prefetch(arow + k0 + 256, 0, 1);  // global_prefetch_b8
    v16bf afrag = load_frag(arow + k0 + koff);
#pragma unroll
    for (int t = 0; t < 4; ++t) {
      v16bf bfrag = load_frag(&buf[(t * 16 + l15) * FC1_LSTRIDE + koff]);
      acc[t] = __builtin_amdgcn_wmma_f32_16x16x32_bf16(
          false, afrag, false, bfrag, (short)0, acc[t], false, false);
    }
    __syncthreads();                            // all reads done before reuse
  }
#else
  // cooperative B staging: 256 threads x 8 halfs = 64 rows x 32 halfs
  int tn = threadIdx.x >> 2;          // 0..63: n-row within panel
  int tk = (threadIdx.x & 3) * 8;     // 0,8,16,24
  const bf16_t* bsrc = wt + (size_t)(nbase + tn) * 3136 + tk;
  for (int kk = 0; kk < 98; ++kk) {
    int k0 = kk * 32;
    bf16_t* buf = lds_b[kk & 1];
    *(v8bf*)(&buf[tn * FC1_LSTRIDE + tk]) = *(const v8bf*)(bsrc + k0);
    __builtin_prefetch(bsrc + k0 + 256, 0, 1);
    __builtin_prefetch(arow + k0 + 256, 0, 1);
    __syncthreads();
    v16bf afrag = load_frag(arow + k0 + koff);
#pragma unroll
    for (int t = 0; t < 4; ++t) {
      v16bf bfrag = load_frag(&buf[(t * 16 + l15) * FC1_LSTRIDE + koff]);
      acc[t] = __builtin_amdgcn_wmma_f32_16x16x32_bf16(
          false, afrag, false, bfrag, (short)0, acc[t], false, false);
    }
    __syncthreads();
  }
#endif

  int rowoff = (lane >= 16) ? 8 : 0;
#pragma unroll
  for (int t = 0; t < 4; ++t) {
    int n = nbase + t * 16 + l15;
    float bb = bias[n];
#pragma unroll
    for (int j = 0; j < 8; ++j) {
      int m = mbase + j + rowoff;
      out[m * 1024 + n] = fmaxf(acc[t][j] + bb, 0.0f);  // bias+relu fused, fp32
    }
  }
}

// ---------- stage 5: fc2 (M=2048, N=10, K=1024) scalar fp32 ----------
__global__ void fc2_kernel(const float* __restrict__ a, const float* __restrict__ w,
                           const float* __restrict__ bias, float* __restrict__ out,
                           int total) {
  int tid = blockIdx.x * blockDim.x + threadIdx.x;
  if (tid >= total) return;           // total = 2048*10
  int o = tid % 10;
  int b = tid / 10;
  const float* ar = a + b * 1024;
  const float* wr = w + o * 1024;
  float s = bias[o];
  for (int k = 0; k < 1024; k += 4) {
    s = fmaf(ar[k],     wr[k],     s);
    s = fmaf(ar[k + 1], wr[k + 1], s);
    s = fmaf(ar[k + 2], wr[k + 2], s);
    s = fmaf(ar[k + 3], wr[k + 3], s);
  }
  out[tid] = s;
}

// ---------- launcher ----------
extern "C" void kernel_launch(void* const* d_in, const int* in_sizes, int n_in,
                              void* d_out, int out_size, void* d_ws, size_t ws_size,
                              hipStream_t stream) {
  const float* x       = (const float*)d_in[0];  // [2048,1,28,28]
  const float* conv1_w = (const float*)d_in[1];  // [32,1,5,5]
  const float* trelu_t = (const float*)d_in[2];  // [32]
  const float* conv2_w = (const float*)d_in[3];  // [64,32,5,5]
  const float* conv2_b = (const float*)d_in[4];  // [64]
  const float* fc1_w   = (const float*)d_in[5];  // [1024,3136]
  const float* fc1_b   = (const float*)d_in[6];  // [1024]
  const float* fc2_w   = (const float*)d_in[7];  // [10,1024]
  const float* fc2_b   = (const float*)d_in[8];  // [10]
  float* out = (float*)d_out;                    // [2048,10]

  const int B = 2048;

  // workspace carve-up (all offsets multiples of 256 bytes)
  char* ws = (char*)d_ws;
  size_t off = 0;
  bf16_t* zbuf    = (bf16_t*)(ws + off); off += 256;                             // 128B zeros
  bf16_t* w2b     = (bf16_t*)(ws + off); off += (size_t)25 * 64 * 32 * 2;        // 102,400
  bf16_t* w1b     = (bf16_t*)(ws + off); off += (size_t)1024 * 3136 * 2;         // 6,422,528
  bf16_t* pooled1 = (bf16_t*)(ws + off); off += (size_t)B * 14 * 14 * 32 * 2;    // 25,690,112
  bf16_t* c2out   = (bf16_t*)(ws + off); off += (size_t)B * 14 * 14 * 64 * 2;    // 51,380,224
  bf16_t* act2    = (bf16_t*)(ws + off); off += (size_t)B * 3136 * 2;            // 12,845,056
  float*  fc1out  = (float*)(ws + off);  off += (size_t)B * 1024 * 4;            // 8,388,608

  // weight converts + zero pad buffer
  zero_pad_kernel<<<1, 64, 0, stream>>>(zbuf);
  cvt_w2_kernel<<<(25 * 64 * 32 + 255) / 256, 256, 0, stream>>>(conv2_w, w2b);
  cvt_w1_kernel<<<(1024 * 3136 + 255) / 256, 256, 0, stream>>>(fc1_w, w1b, 1024 * 3136);

  // conv1 + topk + trelu + pool
  int total1 = B * 14 * 14;  // 401408
  conv1_topk_pool_kernel<<<(total1 + 255) / 256, 256, 0, stream>>>(
      x, conv1_w, trelu_t, pooled1, total1);

  // conv2 WMMA: 25088 M-tiles, 8 waves/block, each wave does all 64 couts
  conv2_wmma_kernel<<<(B * 196 / 16) / 8, 256, 0, stream>>>(
      pooled1, w2b, conv2_b, zbuf, c2out);

  // pool + flatten
  int total2 = B * 64 * 49;  // 6,422,528
  pool2_flatten_kernel<<<(total2 + 255) / 256, 256, 0, stream>>>(c2out, act2, total2);

  // fc1 WMMA: 16 M-groups x 16 N-groups = 256 blocks, 8 waves each
  fc1_wmma_kernel<<<256, 256, 0, stream>>>(act2, w1b, fc1_b, fc1out);

  // fc2
  int total3 = B * 10;
  fc2_kernel<<<(total3 + 255) / 256, 256, 0, stream>>>(fc1out, fc2_w, fc2_b, out, total3);
}
